// CGKN_79714593014065
// MI455X (gfx1250) — matrix operand order repaired and checked by reading, not compiled, not run on hardware
//
#include <hip/hip_runtime.h>
#include <hip/hip_bf16.h>
#include <math.h>

// ---------------------------------------------------------------------------
// CGKN forward, fused for MI455X (gfx1250, wave32, WMMA).
//
// Key idea: the reference materializes g1 [512,256,512] = 268MB, softmaxes it
// and dots with z. We fuse GEMM(32->2048) + softmax + <.,z> per point so the
// 268MB tensor never exists: HBM traffic drops from ~540MB to ~2MB, leaving
// ~4.3 GFLOP of f16 WMMA (K=32 == one v_wmma_f32_16x16x32_f16 per tile).
// z_pred uses full-precision v_wmma_f32_16x16x4_f32.
// ---------------------------------------------------------------------------

typedef _Float16 v16h __attribute__((ext_vector_type(16)));
typedef _Float16 h8   __attribute__((ext_vector_type(8)));
typedef float    v8f  __attribute__((ext_vector_type(8)));
typedef float    v2f  __attribute__((ext_vector_type(2)));

#define LN_EPS 1e-5f
#define PI_F   3.14159265358979323846f

// ---------------- prep: transpose g1.w3 [32,2048] -> f16 [2048,32] ----------
__global__ void __launch_bounds__(256) prep_wg4t(const float* __restrict__ w,
                                                 _Float16* __restrict__ wt) {
    int i = blockIdx.x * 256 + threadIdx.x;      // 65536 elements
    int n = i >> 5;                              // column of original (0..2047)
    int k = i & 31;                              // row of original (0..31)
    wt[i] = (_Float16)w[k * 2048 + n];
}

// ---------------- shared LN + SiLU helper (16-lane shuffle reduction) -------
template <int J, int N>
__device__ __forceinline__ void ln_silu_store(float* x, int l,
                                              const float* __restrict__ G,
                                              const float* __restrict__ E,
                                              float* dst) {
    float s = 0.f, s2 = 0.f;
#pragma unroll
    for (int j = 0; j < J; ++j) { s += x[j]; s2 += x[j] * x[j]; }
#pragma unroll
    for (int m = 8; m >= 1; m >>= 1) {           // stays within 16-lane half
        s  += __shfl_xor(s,  m, 32);
        s2 += __shfl_xor(s2, m, 32);
    }
    const float mean = s * (1.0f / N);
    const float var  = s2 * (1.0f / N) - mean * mean;
    const float rinv = rsqrtf(var + LN_EPS);
#pragma unroll
    for (int j = 0; j < J; ++j) {
        int   o = l + 16 * j;
        float h = (x[j] - mean) * rinv * G[o] + E[o];
        h = h * (1.0f / (1.0f + __expf(-h)));    // SiLU
        dst[o] = h;
    }
}

// ---------------- trunk: 26 -> 128 -> 64 -> 32 (LN+SiLU each) ---------------
// wave w handles points 2w (lanes 0-15) and 2w+1 (lanes 16-31).
__device__ __forceinline__ void mlp_trunk(
    const float* __restrict__ W1, const float* __restrict__ B1,
    const float* __restrict__ G1, const float* __restrict__ E1,
    const float* __restrict__ W2, const float* __restrict__ B2,
    const float* __restrict__ G2, const float* __restrict__ E2,
    const float* __restrict__ W3, const float* __restrict__ B3,
    const float* __restrict__ G3, const float* __restrict__ E3,
    const float* peL, float* h1, float* h2, float* h3) {
    const int lane = threadIdx.x & 31;
    const int wv   = threadIdx.x >> 5;
    const int l    = lane & 15;
    const int p    = wv * 2 + (lane >> 4);

    {   // layer 1: 26 -> 128, 8 outputs/lane
        float x[8];
#pragma unroll
        for (int j = 0; j < 8; ++j) {
            int   o   = l + 16 * j;
            float acc = B1[o];
#pragma unroll
            for (int k = 0; k < 26; ++k) acc += peL[p * 26 + k] * W1[k * 128 + o];
            x[j] = acc;
        }
        ln_silu_store<8, 128>(x, l, G1, E1, h1 + p * 128);
    }
    __syncthreads();
    {   // layer 2: 128 -> 64, 4 outputs/lane
        float x[4];
#pragma unroll
        for (int j = 0; j < 4; ++j) {
            int   o   = l + 16 * j;
            float acc = B2[o];
            for (int k = 0; k < 128; ++k) acc += h1[p * 128 + k] * W2[k * 64 + o];
            x[j] = acc;
        }
        ln_silu_store<4, 64>(x, l, G2, E2, h2 + p * 64);
    }
    __syncthreads();
    {   // layer 3: 64 -> 32, 2 outputs/lane
        float x[2];
#pragma unroll
        for (int j = 0; j < 2; ++j) {
            int   o   = l + 16 * j;
            float acc = B3[o];
            for (int k = 0; k < 64; ++k) acc += h2[p * 64 + k] * W3[k * 32 + o];
            x[j] = acc;
        }
        ln_silu_store<2, 32>(x, l, G3, E3, h3 + p * 32);
    }
    __syncthreads();
}

struct FusedArgs {
    const float *u1, *z;
    const float *fw0, *fw1, *fw2, *fw3, *fb0, *fb1, *fb2, *fb3;
    const float *fg0, *fg1, *fg2, *fe0, *fe1, *fe2;
    const float *gw0, *gw1, *gw2, *gb0, *gb1, *gb2, *gb3;
    const float *gg0, *gg1, *gg2, *ge0, *ge1, *ge2;
    const _Float16 *wg4t;    // g1.w3 transposed f16 [2048][32]
    float *out_u1;           // [512*64*4]
};

// ---------------- fused per-point kernel ------------------------------------
// 16 points per workgroup (all share the same batch row b), 256 threads.
__global__ void __launch_bounds__(256) fused_kernel(FusedArgs a) {
    __shared__ float peL[16 * 26];
    __shared__ float h1[16 * 128];
    __shared__ float h2[16 * 64];
    __shared__ float h3[16 * 32];
    __shared__ __align__(16) _Float16 h3h[16 * 32];
    __shared__ float f1s[16 * 4];
    __shared__ float zsh[512];
    __shared__ float pnum[8][16];
    __shared__ float pden[8][16];

    const int tid   = threadIdx.x;
    const int gbase = blockIdx.x * 16;           // global point base
    const int brow  = gbase >> 6;                // shared batch index

    // stage z row
    for (int i = tid; i < 512; i += 256) zsh[i] = a.z[brow * 512 + i];

    // positional encoding (each thread recomputes its atan2 -> no extra sync)
    if (tid < 192) {
        int   p = tid / 12, r = tid % 12, axis = r / 6, f = r % 6;
        int   gpt = gbase + p;
        float ux  = a.u1[gpt * 4 + axis * 2];
        float uy  = a.u1[gpt * 4 + axis * 2 + 1];
        float ang = atan2f(uy, ux);
        if (f == 0) peL[p * 26 + axis] = ang;
        float sv, cv;
        __sincosf(ang * ((float)(1 << f) * PI_F), &sv, &cv);
        peL[p * 26 + 2 + axis * 12 + f]     = sv;
        peL[p * 26 + 2 + axis * 12 + 6 + f] = cv;
    }
    __syncthreads();

    // ---- f1 trunk + 32->4 head ----
    mlp_trunk(a.fw0, a.fb0, a.fg0, a.fe0, a.fw1, a.fb1, a.fg1, a.fe1,
              a.fw2, a.fb2, a.fg2, a.fe2, peL, h1, h2, h3);
    if (tid < 64) {
        int   p = tid >> 2, c = tid & 3;
        float acc = a.fb3[c];
#pragma unroll
        for (int k = 0; k < 32; ++k) acc += h3[p * 32 + k] * a.fw3[k * 4 + c];
        f1s[p * 4 + c] = acc;
    }
    __syncthreads();

    // ---- g1 trunk ----
    mlp_trunk(a.gw0, a.gb0, a.gg0, a.ge0, a.gw1, a.gb1, a.gg1, a.ge1,
              a.gw2, a.gb2, a.gg2, a.ge2, peL, h1, h2, h3);
    for (int i = tid; i < 512; i += 256) h3h[i] = (_Float16)h3[i];
    __syncthreads();

    // ---- big head 32->2048 via WMMA, fused softmax-dot with z ----
    // wave wv covers columns [wv*256, wv*256+256); softmax group = 512 cols
    // = waves {2g, 2g+1}. Logits ~N(0,1) => exp without max-sub is safe.
    const int lane = tid & 31;
    const int wv   = tid >> 5;
    const int nloc = lane & 15;
    const int half = lane >> 4;

    // A (16x32 f16): lane half h holds K {h*8..h*8+7} and {16+h*8..16+h*8+7}
    union { v16h v; h8 h[2]; } au;
    {
        const int kb = half * 8;
        au.h[0] = *(const h8*)(h3h + nloc * 32 + kb);
        au.h[1] = *(const h8*)(h3h + nloc * 32 + 16 + kb);
    }

    v8f num = {}; v8f den = {};
    const int colbase = wv * 256;
#pragma unroll 4
    for (int t = 0; t < 16; ++t) {
        const int colg = colbase + t * 16 + nloc;   // this lane's column
        // B (32x16 f16): lane holds col nloc, K {half*8..half*8+7} and {16+half*8..16+half*8+7}
        union { v16h v; h8 h[2]; } bu;
        const _Float16* bp = a.wg4t + colg * 32 + half * 8;
        bu.h[0] = *(const h8*)(bp);
        bu.h[1] = *(const h8*)(bp + 16);

        v8f cz = {};
        v8f acc = __builtin_amdgcn_wmma_f32_16x16x32_f16(
            false, au.v, false, bu.v, (short)0, cz, false, false);

        const float bias = a.gb3[colg];
        const float zval = zsh[colg & 511];
#pragma unroll
        for (int v = 0; v < 8; ++v) {
            float e = __expf(acc[v] + bias);
            num[v] += e * zval;
            den[v] += e;
        }
    }
    // reduce across the 16 columns held by lanes of this half
#pragma unroll
    for (int m = 8; m >= 1; m >>= 1) {
#pragma unroll
        for (int v = 0; v < 8; ++v) {
            num[v] += __shfl_xor(num[v], m, 32);
            den[v] += __shfl_xor(den[v], m, 32);
        }
    }
    if (nloc == 0) {
#pragma unroll
        for (int v = 0; v < 8; ++v) {
            pnum[wv][half * 8 + v] = num[v];
            pden[wv][half * 8 + v] = den[v];
        }
    }
    __syncthreads();

    // combine the two waves of each softmax group; write u1_pred
    if (tid < 64) {
        int   p = tid & 15, c = tid >> 4;
        float N = pnum[2 * c][p] + pnum[2 * c + 1][p];
        float D = pden[2 * c][p] + pden[2 * c + 1][p];
        a.out_u1[(gbase + p) * 4 + c] = f1s[p * 4 + c] + N / D;
    }
}

// ---------------- z_pred = f2 + z @ g2^T  (fp32 WMMA 16x16x4) ---------------
__global__ void __launch_bounds__(256) zpred_kernel(const float* __restrict__ z,
                                                    const float* __restrict__ g2,
                                                    const float* __restrict__ f2,
                                                    float* __restrict__ out) {
    const int lane = threadIdx.x & 31;
    const int wv   = threadIdx.x >> 5;
    const int tile = blockIdx.x * 8 + wv;        // 1024 tiles total
    const int tm   = tile >> 5, tn = tile & 31;
    const int nloc = lane & 15, half = lane >> 4;
    const int bm   = tm * 16 + nloc;             // A row (batch)
    const int ng   = tn * 16 + nloc;             // B col (output feature i)

    const float* arow = z  + bm * 512 + 2 * half;   // A: M=nloc, K=2*half+{0,1}
    const float* brow = g2 + ng * 512 + 2 * half;   // B[k][n] = g2[n][k]

    v8f c = {};
#pragma unroll 8
    for (int kk = 0; kk < 512; kk += 4) {
        v2f av = { arow[kk], arow[kk + 1] };
        v2f bv = { brow[kk], brow[kk + 1] };
        c = __builtin_amdgcn_wmma_f32_16x16x4_f32(
            false, av, false, bv, (short)0, c, false, false);
    }
    const float bias = f2[ng];
#pragma unroll
    for (int v = 0; v < 8; ++v) {
        int row = tm * 16 + half * 8 + v;        // D: lane col = nloc, rows half*8+v
        out[row * 512 + ng] = c[v] + bias;
    }
}

// ---------------------------------------------------------------------------
extern "C" void kernel_launch(void* const* d_in, const int* in_sizes, int n_in,
                              void* d_out, int out_size, void* d_ws, size_t ws_size,
                              hipStream_t stream) {
    (void)in_sizes; (void)n_in; (void)out_size; (void)ws_size;
    // setup_inputs insertion order; nested _mlp_params dicts in insertion
    // order {w[4], b[4], g[3], be[3]}.
    const float* u1 = (const float*)d_in[0];
    const float* z  = (const float*)d_in[1];

    FusedArgs fa;
    fa.u1 = u1; fa.z = z;
    fa.fw0 = (const float*)d_in[2];  fa.fw1 = (const float*)d_in[3];
    fa.fw2 = (const float*)d_in[4];  fa.fw3 = (const float*)d_in[5];
    fa.fb0 = (const float*)d_in[6];  fa.fb1 = (const float*)d_in[7];
    fa.fb2 = (const float*)d_in[8];  fa.fb3 = (const float*)d_in[9];
    fa.fg0 = (const float*)d_in[10]; fa.fg1 = (const float*)d_in[11];
    fa.fg2 = (const float*)d_in[12];
    fa.fe0 = (const float*)d_in[13]; fa.fe1 = (const float*)d_in[14];
    fa.fe2 = (const float*)d_in[15];
    fa.gw0 = (const float*)d_in[16]; fa.gw1 = (const float*)d_in[17];
    fa.gw2 = (const float*)d_in[18];
    const float* gw3 = (const float*)d_in[19];   // [32,2048]
    fa.gb0 = (const float*)d_in[20]; fa.gb1 = (const float*)d_in[21];
    fa.gb2 = (const float*)d_in[22]; fa.gb3 = (const float*)d_in[23];
    fa.gg0 = (const float*)d_in[24]; fa.gg1 = (const float*)d_in[25];
    fa.gg2 = (const float*)d_in[26];
    fa.ge0 = (const float*)d_in[27]; fa.ge1 = (const float*)d_in[28];
    fa.ge2 = (const float*)d_in[29];
    const float* f2v = (const float*)d_in[30];   // [512,1]
    const float* g2m = (const float*)d_in[31];   // [512,512]

    _Float16* wg4t = (_Float16*)d_ws;            // 2048*32*2 = 128KB scratch
    fa.wg4t  = wg4t;
    fa.out_u1 = (float*)d_out;                   // first 131072 floats
    float* out_z = (float*)d_out + 512 * 64 * 4; // next 262144 floats

    prep_wg4t<<<256, 256, 0, stream>>>(gw3, wg4t);
    fused_kernel<<<2048, 256, 0, stream>>>(fa);
    zpred_kernel<<<128, 256, 0, stream>>>(z, g2m, f2v, out_z);
}
